// SelfModifyingTitans_23991687315642
// MI455X (gfx1250) — matrix operand order repaired and checked by reading, not compiled, not run on hardware
//
#include <hip/hip_runtime.h>
#include <hip/hip_bf16.h>
#include <math.h>

// ---------------------------------------------------------------------------
// SelfModifyingTitans for MI455X (gfx1250, wave32, WMMA)
//   B=4, S=2048, D=512, H=8, DK=DV=64, DH=256, KCONV=4
// ---------------------------------------------------------------------------

#define Bsz   4
#define Ssz   2048
#define Dsz   512
#define Hsz   8
#define DKsz  64
#define DVsz  64
#define DHsz  256
#define Mrows (Bsz * Ssz)          // 8192 tokens
#define LN_EPS 1e-5f

typedef __attribute__((ext_vector_type(16))) __bf16 bf16x16;
typedef __attribute__((ext_vector_type(8)))  float  f32x8;
typedef __attribute__((ext_vector_type(4)))  unsigned int u32x4;

union FragU { u32x4 u[2]; bf16x16 v; };

// ---------------------------------------------------------------------------
// f32 -> bf16 conversion (weights & activations)
// ---------------------------------------------------------------------------
__global__ void cvt_bf16_kernel(const float* __restrict__ src,
                                __bf16* __restrict__ dst, int n) {
  int i = blockIdx.x * blockDim.x + threadIdx.x;
  if (i < n) dst[i] = (__bf16)src[i];
}

// ---------------------------------------------------------------------------
// Depthwise causal conv(K=4) + bias + residual; emits f32 and bf16 copies
// ---------------------------------------------------------------------------
__global__ void conv_kernel(const float* __restrict__ x,
                            const float* __restrict__ w,   // [D,1,4]
                            const float* __restrict__ cb,  // [D]
                            float* __restrict__ Xf, __bf16* __restrict__ Xb) {
  int i = blockIdx.x * blockDim.x + threadIdx.x;
  if (i >= Bsz * Ssz * Dsz) return;
  int d = i % Dsz;
  int t = (i / Dsz) % Ssz;
  int b = i / (Dsz * Ssz);
  const float* wd = w + d * 4;
  float acc = cb[d];
#pragma unroll
  for (int j = 0; j < 4; ++j) {
    int ts = t - 3 + j;
    if (ts >= 0)
      acc = fmaf(wd[j], x[((size_t)b * Ssz + ts) * Dsz + d], acc);
  }
  float r = x[i] + acc;
  Xf[i] = r;
  Xb[i] = (__bf16)r;
}

// ---------------------------------------------------------------------------
// WMMA bf16 GEMM: C[M,N] = epilogue( A[M,K] @ W[N,K]^T )
//   ACT: 0 none, 1 silu      RES: += Res[M,N] (f32)     BIAS: += Bias[N]
//   OUTM: 0 f32, 1 bf16, 2 both
// Register-blocked: each wave computes a 32(M) x 64(N) macro-tile
//   = 2 A-frags x 4 B-frags -> 8 accumulators, 8 WMMAs per 12 b128 loads.
// A frag: lane(m=l&15,g=l>>4) holds k = k0+8g+{0..7} and k0+8g+16+{0..7}.
// B frag: lane(n=l&15,g=l>>4) holds k = k0+8g+{0..7} and k0+8g+16+{0..7}.
// ---------------------------------------------------------------------------
template <int ACT, int RES, int BIAS, int OUTM>
__global__ void gemm_bf16_kernel(const __bf16* __restrict__ A,
                                 const __bf16* __restrict__ W,
                                 const float* __restrict__ Res,
                                 const float* __restrict__ Bias,
                                 float* __restrict__ Cf,
                                 __bf16* __restrict__ Cb,
                                 int M, int N, int K) {
  const int wave = threadIdx.x >> 5;
  const int lane = threadIdx.x & 31;
  const int mTiles = M >> 5;                 // 32 rows per macro-tile
  const int nTiles = N >> 6;                 // 64 cols per macro-tile
  const int wt = blockIdx.x * (blockDim.x >> 5) + wave;
  if (wt >= mTiles * nTiles) return;
  const int m0 = (wt % mTiles) << 5;
  const int n0 = (wt / mTiles) << 6;

  const int mr  = lane & 15;
  const int grp = lane >> 4;

  const __bf16* arow = A + (size_t)(m0 + mr) * K + 8 * grp;   // +16*K for mi=1
  const __bf16* brow = W + (size_t)(n0 + mr) * K + 8 * grp;   // +16*K per ni
  const size_t rstep = (size_t)16 * K;

  f32x8 acc[2][4] = {};
  for (int k0 = 0; k0 < K; k0 += 32) {
    FragU af[2], bf[4];
#pragma unroll
    for (int mi = 0; mi < 2; ++mi) {
      const __bf16* p = arow + mi * rstep + k0;
      af[mi].u[0] = *(const u32x4*)(p);
      af[mi].u[1] = *(const u32x4*)(p + 16);
      __builtin_prefetch(p + 32, 0, 3);       // near-cache global_prefetch_b8
    }
#pragma unroll
    for (int ni = 0; ni < 4; ++ni) {
      const __bf16* p = brow + ni * rstep + k0;
      bf[ni].u[0] = *(const u32x4*)(p);
      bf[ni].u[1] = *(const u32x4*)(p + 16);
    }
#pragma unroll
    for (int mi = 0; mi < 2; ++mi)
#pragma unroll
      for (int ni = 0; ni < 4; ++ni)
        acc[mi][ni] = __builtin_amdgcn_wmma_f32_16x16x32_bf16(
            /*neg_a=*/false, af[mi].v, /*neg_b=*/false, bf[ni].v,
            /*c_mod=*/(short)0, acc[mi][ni],
            /*reuse_a=*/false, /*reuse_b=*/false);
  }

  // C layout per 16x16 tile: lane col = (lane&15); VGPR r -> row r + 8*(lane>>4)
#pragma unroll
  for (int mi = 0; mi < 2; ++mi) {
#pragma unroll
    for (int ni = 0; ni < 4; ++ni) {
      const int ncol = n0 + ni * 16 + (lane & 15);
#pragma unroll
      for (int r = 0; r < 8; ++r) {
        const int mrow = m0 + mi * 16 + r + 8 * grp;
        float c = acc[mi][ni][r];
        if (RES)  c += Res[(size_t)mrow * N + ncol];
        if (BIAS) c += Bias[ncol];
        if (ACT)  c = c / (1.f + __expf(-c));     // silu
        const size_t o = (size_t)mrow * N + ncol;
        if (OUTM == 0 || OUTM == 2) Cf[o] = c;
        if (OUTM == 1 || OUTM == 2) Cb[o] = (__bf16)c;
      }
    }
  }
}

// ---------------------------------------------------------------------------
// LayerNorm over the 512-wide channel dim, in place; one wave per row
// ---------------------------------------------------------------------------
__global__ void ln512_kernel(float* __restrict__ x,
                             const float* __restrict__ g,
                             const float* __restrict__ b, int rows) {
  const int wave = threadIdx.x >> 5;
  const int lane = threadIdx.x & 31;
  const int row = blockIdx.x * (blockDim.x >> 5) + wave;
  if (row >= rows) return;
  float* p = x + (size_t)row * 512;
  float vals[16];
  float s = 0.f;
#pragma unroll
  for (int i = 0; i < 16; ++i) { vals[i] = p[lane + 32 * i]; s += vals[i]; }
#pragma unroll
  for (int m = 16; m >= 1; m >>= 1) s += __shfl_xor(s, m, 32);
  const float mu = s * (1.f / 512.f);
  float vs = 0.f;
#pragma unroll
  for (int i = 0; i < 16; ++i) { float d = vals[i] - mu; vs = fmaf(d, d, vs); }
#pragma unroll
  for (int m = 16; m >= 1; m >>= 1) vs += __shfl_xor(vs, m, 32);
  const float inv = rsqrtf(vs * (1.f / 512.f) + LN_EPS);
#pragma unroll
  for (int i = 0; i < 16; ++i) {
    int c = lane + 32 * i;
    p[c] = (vals[i] - mu) * inv * g[c] + b[c];
  }
}

// ---------------------------------------------------------------------------
// eta/alpha heads: out[t,h] = act( hid[t,:256] . W1[h,:256] )
//   ACT: 0 -> softplus()*0.1 ; 1 -> sigmoid()
// ---------------------------------------------------------------------------
template <int ACT>
__global__ void head_kernel(const __bf16* __restrict__ hid,
                            const float* __restrict__ W1,  // [H, 256]
                            float* __restrict__ out) {
  int idx = blockIdx.x * blockDim.x + threadIdx.x;
  if (idx >= Mrows * Hsz) return;
  const int t = idx >> 3, h = idx & 7;
  const __bf16* hp = hid + (size_t)t * DHsz;
  const float* wp = W1 + h * DHsz;
  float s = 0.f;
#pragma unroll 8
  for (int j = 0; j < DHsz; ++j) s = fmaf((float)hp[j], wp[j], s);
  if (ACT == 0) {
    float sp = (s > 20.f) ? s : log1pf(__expf(s));
    out[idx] = 0.1f * sp;
  } else {
    out[idx] = 1.f / (1.f + __expf(-s));
  }
}

// ---------------------------------------------------------------------------
// Recurrent scan. Grid = B*H = 32 blocks, 128 threads.
// Thread (v = tid>>1, half = tid&1) owns mem[v][half*32 .. +31] in VGPRs.
// Rank-1 update form: mem = a*mem - e*(2*(mem.kn) - vh) kn^T ; y = mem.q
// Lane pairs (2v, 2v+1) are in the same wave32 -> combine with shfl_xor(.,1).
// ---------------------------------------------------------------------------
__global__ void scan_kernel(const float* __restrict__ q,
                            const float* __restrict__ k,
                            const float* __restrict__ vh,
                            const float* __restrict__ eta,
                            const float* __restrict__ alpha,
                            float* __restrict__ Y) {
  const int bh = blockIdx.x;
  const int b = bh >> 3, h = bh & 7;
  const int tid = threadIdx.x;
  const int v = tid >> 1, half = tid & 1;

  __shared__ float shq[64], shk[64], shvh[64], shea[2];

  float mem[32];
#pragma unroll
  for (int j = 0; j < 32; ++j) mem[j] = 0.f;

  for (int t = 0; t < Ssz; ++t) {
    const size_t base = ((size_t)b * Ssz + t) * (Hsz * DVsz) + h * DVsz;
    if (tid < 64) {
      shq[tid]  = q[base + tid];
      shk[tid]  = k[base + tid];
      shvh[tid] = vh[base + tid];
    }
    if (tid == 64) {
      const size_t eb = ((size_t)b * Ssz + t) * Hsz + h;
      shea[0] = eta[eb];
      shea[1] = alpha[eb];
    }
    __syncthreads();

    const float* qh = shq + half * 32;
    const float* kh = shk + half * 32;
    float y = 0.f, mk = 0.f, kk = 0.f;
#pragma unroll
    for (int j = 0; j < 32; ++j) {
      const float m = mem[j];
      y  = fmaf(m, qh[j], y);
      mk = fmaf(m, kh[j], mk);
      kk = fmaf(kh[j], kh[j], kk);
    }
    y  += __shfl_xor(y, 1, 32);
    mk += __shfl_xor(mk, 1, 32);
    kk += __shfl_xor(kk, 1, 32);

    const float inv = 1.f / fmaxf(sqrtf(kk), 1e-12f);
    mk *= inv;                                   // mem . kn  (row v)
    const float e = shea[0], a = shea[1];
    const float coef = e * (2.f * mk - shvh[v]); // e*(mk + err)
#pragma unroll
    for (int j = 0; j < 32; ++j)
      mem[j] = fmaf(a, mem[j], -coef * (kh[j] * inv));

    if (half == 0) Y[base + v] = y;
    __syncthreads();
  }
}

// ---------------------------------------------------------------------------
// Host-side orchestration
// ---------------------------------------------------------------------------
extern "C" void kernel_launch(void* const* d_in, const int* in_sizes, int n_in,
                              void* d_out, int out_size, void* d_ws, size_t ws_size,
                              hipStream_t stream) {
  (void)in_sizes; (void)n_in; (void)out_size; (void)ws_size;

  const float* x       = (const float*)d_in[0];
  const float* conv_w  = (const float*)d_in[1];
  const float* conv_b  = (const float*)d_in[2];
  const float* Wq      = (const float*)d_in[3];
  const float* Mk_W2   = (const float*)d_in[4];
  const float* Mk_W1   = (const float*)d_in[5];
  const float* Mv_W2   = (const float*)d_in[6];
  const float* Mv_W1   = (const float*)d_in[7];
  const float* Meta_W2 = (const float*)d_in[8];
  const float* Meta_W1 = (const float*)d_in[9];
  const float* Mal_W2  = (const float*)d_in[10];
  const float* Mal_W1  = (const float*)d_in[11];
  const float* Mvh_W2  = (const float*)d_in[12];
  const float* Mvh_W1  = (const float*)d_in[13];
  const float* normk_g = (const float*)d_in[14];
  const float* normk_b = (const float*)d_in[15];
  const float* normq_g = (const float*)d_in[16];
  const float* normq_b = (const float*)d_in[17];
  const float* out_W   = (const float*)d_in[18];
  const float* out_b   = (const float*)d_in[19];
  float* out = (float*)d_out;

  char* ws = (char*)d_ws;
  // ---- workspace layout (bytes) ----
  float*  Xf  = (float*) (ws + 0);                         // 16 MB
  __bf16* Xb  = (__bf16*)(ws + 16777216);                  //  8 MB
  __bf16* Hid = (__bf16*)(ws + 25165824);                  //  4 MB
  float*  kf  = (float*) (ws + 29360128);                  // 16 MB
  float*  qf  = (float*) (ws + 46137344);                  // 16 MB
  float*  vf  = (float*) (ws + 62914560);                  // 16 MB
  __bf16* vb  = (__bf16*)(ws + 79691776);                  //  8 MB
  float*  vhf = (float*) (ws + 88080384);                  // 16 MB
  float*  etaB= (float*) (ws + 104857600);                 // 256 KB
  float*  alB = (float*) (ws + 105119744);                 // 256 KB
  __bf16* WB  = (__bf16*)(ws + 105381888);                 //  3 MB bf16 weights
  // scan output aliases dead post-conv buffers
  float*  Yf  = Xf;
  __bf16* Yb  = Xb;

  __bf16* Wqb   = WB;
  __bf16* MkW2b = WB + 262144;
  __bf16* MkW1b = WB + 393216;
  __bf16* MvW2b = WB + 524288;
  __bf16* MvW1b = WB + 655360;
  __bf16* MeW2b = WB + 786432;
  __bf16* MaW2b = WB + 917504;
  __bf16* MhW2b = WB + 1048576;
  __bf16* MhW1b = WB + 1179648;
  __bf16* outWb = WB + 1310720;

  const int TPB = 256;
  // ---- weight conversions (f32 -> bf16) ----
  cvt_bf16_kernel<<<(262144 + TPB - 1) / TPB, TPB, 0, stream>>>(Wq,     Wqb,   262144);
  cvt_bf16_kernel<<<(131072 + TPB - 1) / TPB, TPB, 0, stream>>>(Mk_W2,  MkW2b, 131072);
  cvt_bf16_kernel<<<(131072 + TPB - 1) / TPB, TPB, 0, stream>>>(Mk_W1,  MkW1b, 131072);
  cvt_bf16_kernel<<<(131072 + TPB - 1) / TPB, TPB, 0, stream>>>(Mv_W2,  MvW2b, 131072);
  cvt_bf16_kernel<<<(131072 + TPB - 1) / TPB, TPB, 0, stream>>>(Mv_W1,  MvW1b, 131072);
  cvt_bf16_kernel<<<(131072 + TPB - 1) / TPB, TPB, 0, stream>>>(Meta_W2,MeW2b, 131072);
  cvt_bf16_kernel<<<(131072 + TPB - 1) / TPB, TPB, 0, stream>>>(Mal_W2, MaW2b, 131072);
  cvt_bf16_kernel<<<(131072 + TPB - 1) / TPB, TPB, 0, stream>>>(Mvh_W2, MhW2b, 131072);
  cvt_bf16_kernel<<<(131072 + TPB - 1) / TPB, TPB, 0, stream>>>(Mvh_W1, MhW1b, 131072);
  cvt_bf16_kernel<<<(262144 + TPB - 1) / TPB, TPB, 0, stream>>>(out_W,  outWb, 262144);

  // ---- conv + residual ----
  conv_kernel<<<(Bsz * Ssz * Dsz + TPB - 1) / TPB, TPB, 0, stream>>>(
      x, conv_w, conv_b, Xf, Xb);

  // 32x64 macro-tiles, 8 waves per block
  auto gemm_blocks = [](int M, int N) { return ((M >> 5) * (N >> 6) + 7) / 8; };

  // ---- k = LN(MLP_k(x)) ----
  gemm_bf16_kernel<1, 0, 0, 1><<<gemm_blocks(Mrows, DHsz), TPB, 0, stream>>>(
      Xb, MkW2b, nullptr, nullptr, nullptr, Hid, Mrows, DHsz, Dsz);
  gemm_bf16_kernel<0, 0, 0, 0><<<gemm_blocks(Mrows, 512), TPB, 0, stream>>>(
      Hid, MkW1b, nullptr, nullptr, kf, nullptr, Mrows, 512, DHsz);
  ln512_kernel<<<(Mrows + 7) / 8, TPB, 0, stream>>>(kf, normk_g, normk_b, Mrows);

  // ---- v = MLP_v(x)  (f32 + bf16 copies) ----
  gemm_bf16_kernel<1, 0, 0, 1><<<gemm_blocks(Mrows, DHsz), TPB, 0, stream>>>(
      Xb, MvW2b, nullptr, nullptr, nullptr, Hid, Mrows, DHsz, Dsz);
  gemm_bf16_kernel<0, 0, 0, 2><<<gemm_blocks(Mrows, 512), TPB, 0, stream>>>(
      Hid, MvW1b, nullptr, nullptr, vf, vb, Mrows, 512, DHsz);

  // ---- q = LN(x @ Wq^T) ----
  gemm_bf16_kernel<0, 0, 0, 0><<<gemm_blocks(Mrows, 512), TPB, 0, stream>>>(
      Xb, Wqb, nullptr, nullptr, qf, nullptr, Mrows, 512, Dsz);
  ln512_kernel<<<(Mrows + 7) / 8, TPB, 0, stream>>>(qf, normq_g, normq_b, Mrows);

  // ---- eta = softplus(MLP_eta(x)) * 0.1 ----
  gemm_bf16_kernel<1, 0, 0, 1><<<gemm_blocks(Mrows, DHsz), TPB, 0, stream>>>(
      Xb, MeW2b, nullptr, nullptr, nullptr, Hid, Mrows, DHsz, Dsz);
  head_kernel<0><<<(Mrows * Hsz + TPB - 1) / TPB, TPB, 0, stream>>>(Hid, Meta_W1, etaB);

  // ---- alpha = sigmoid(MLP_alpha(x)) ----
  gemm_bf16_kernel<1, 0, 0, 1><<<gemm_blocks(Mrows, DHsz), TPB, 0, stream>>>(
      Xb, MaW2b, nullptr, nullptr, nullptr, Hid, Mrows, DHsz, Dsz);
  head_kernel<1><<<(Mrows * Hsz + TPB - 1) / TPB, TPB, 0, stream>>>(Hid, Mal_W1, alB);

  // ---- v_hat = v + MLP_vh(v) ----
  gemm_bf16_kernel<1, 0, 0, 1><<<gemm_blocks(Mrows, DHsz), TPB, 0, stream>>>(
      vb, MhW2b, nullptr, nullptr, nullptr, Hid, Mrows, DHsz, 512);
  gemm_bf16_kernel<0, 1, 0, 0><<<gemm_blocks(Mrows, 512), TPB, 0, stream>>>(
      Hid, MhW1b, vf, nullptr, vhf, nullptr, Mrows, 512, DHsz);

  // ---- recurrent scan (32 chains) ----
  scan_kernel<<<Bsz * Hsz, 128, 0, stream>>>(qf, kf, vhf, etaB, alB, Yf);

  // ---- final projection: out = Y @ out_W^T + out_b ----
  cvt_bf16_kernel<<<(Mrows * 512 + TPB - 1) / TPB, TPB, 0, stream>>>(Yf, Yb, Mrows * 512);
  gemm_bf16_kernel<0, 0, 1, 0><<<gemm_blocks(Mrows, Dsz), TPB, 0, stream>>>(
      Yb, outWb, nullptr, out_b, out, nullptr, Mrows, Dsz, 512);
}
